// Encoder_37203006718635
// MI455X (gfx1250) — compile-verified, matrix operand order checked by reading, not run on hardware
//
#include <hip/hip_runtime.h>
#include <hip/hip_bf16.h>
#include <math.h>

// ---------------------------------------------------------------------------
// CDNA5 (gfx1250) WMMA bf16 implementation of the bi-GRU + segment-GRU encoder.
// All input projections are hoisted out of the recurrences into big WMMA GEMMs;
// recurrences run 256 independent chains (one workgroup each) doing
// h(8x256) @ Wh^T(256->768) per step with v_wmma_f32_16x16x32_bf16.
// ---------------------------------------------------------------------------

typedef __bf16 bf16_t;
typedef __attribute__((ext_vector_type(16))) __bf16 v16bf;
typedef __attribute__((ext_vector_type(8)))  __bf16 v8bf;
typedef __attribute__((ext_vector_type(8)))  float  v8f;

#define SEQ_N 128
#define BATCH 8
#define HID_H 256
#define SEG_S 256
#define SEG_T 32
#define G3    768    // 3*H == 3*S
#define IN2H  512    // 2*H
#define NPAIR 8256   // N*(N+1)/2

static __device__ __forceinline__ v8f wmma_bf16(v16bf a, v16bf b, v8f c) {
  // D = A(16x32) * B(32x16) + C, fp32 accum
  return __builtin_amdgcn_wmma_f32_16x16x32_bf16(false, a, false, b, (short)0, c,
                                                 false, false);
}

// Load one 16x32 bf16 fragment from a row-major matrix (row stride = ldk elems).
// A-layout per ISA 7.12.2: lane L (<16): M=L, K={k0..k0+7, k0+16..k0+23};
// lane L+16: same M, K shifted by +8.  B fragments use the same layout applied
// to W (Nout x K) row-major, i.e. B^T.
static __device__ __forceinline__ v16bf load_frag(const bf16_t* __restrict__ base,
                                                  int ldk, int row0, int k0, int lane) {
  int r  = row0 + (lane & 15);
  int kb = k0 + ((lane & 16) ? 8 : 0);
  const bf16_t* p = base + (size_t)r * (size_t)ldk + kb;
  v8bf lo = *(const v8bf*)(p);
  v8bf hi = *(const v8bf*)(p + 16);
  return __builtin_shufflevector(lo, hi, 0,1,2,3,4,5,6,7,8,9,10,11,12,13,14,15);
}

static __device__ __forceinline__ float sigm(float x) {
  return 1.0f / (1.0f + __expf(-x));
}

// gh[8x768] = hB[16x256(bf16, rows 8..15 zero)] @ Wh^T  (Wh is 768x256 bf16)
// 16 waves, each owns 3 of the 48 N-tiles.
static __device__ __forceinline__ void gh_gemm(const bf16_t* __restrict__ hB,
                                               const bf16_t* __restrict__ Wh,
                                               float* __restrict__ gh,
                                               int wave, int lane) {
  const int n0 = wave * 48;
  v8f a0 = {0.f,0.f,0.f,0.f,0.f,0.f,0.f,0.f};
  v8f a1 = a0, a2 = a0;
  #pragma unroll
  for (int k0 = 0; k0 < 256; k0 += 32) {
    v16bf a  = load_frag(hB, 256, 0, k0, lane);
    v16bf b0 = load_frag(Wh, 256, n0,      k0, lane);
    v16bf b1 = load_frag(Wh, 256, n0 + 16, k0, lane);
    v16bf b2 = load_frag(Wh, 256, n0 + 32, k0, lane);
    a0 = wmma_bf16(a, b0, a0);
    a1 = wmma_bf16(a, b1, a1);
    a2 = wmma_bf16(a, b2, a2);
  }
  // C/D layout: lane<16 holds rows M=0..7 (the real batch rows), col N = lane.
  if (lane < 16) {
    #pragma unroll
    for (int r = 0; r < 8; ++r) {
      gh[r * G3 + n0      + lane] = a0[r];
      gh[r * G3 + n0 + 16 + lane] = a1[r];
      gh[r * G3 + n0 + 32 + lane] = a2[r];
    }
  }
}

// ---------------- utility kernels ----------------

__global__ void zero_kernel(float4* __restrict__ p, int n4) {
  int i = blockIdx.x * 256 + threadIdx.x;
  if (i < n4) p[i] = make_float4(0.f, 0.f, 0.f, 0.f);
}

__global__ void f2bf_kernel(const float* __restrict__ src, bf16_t* __restrict__ dst, int n) {
  int i = blockIdx.x * 256 + threadIdx.x;
  if (i < n) dst[i] = (bf16_t)src[i];
}

__global__ void embed_kernel(const int* __restrict__ tokens,
                             const float* __restrict__ emb,
                             bf16_t* __restrict__ Xb) {
  int row = blockIdx.x;           // t*8 + b, 0..1023
  int e   = threadIdx.x;          // 0..255
  int tok = tokens[row];
  Xb[(size_t)row * 256 + e] = (bf16_t)emb[(size_t)tok * 256 + e];
}

// C(M x Nout) = A(M x K, bf16) @ W(Nout x K, bf16)^T + bias.  One 16x16 tile/wave.
__global__ __launch_bounds__(256) void gemm_bias_kernel(const bf16_t* __restrict__ A,
                                                        const bf16_t* __restrict__ W,
                                                        const float* __restrict__ bias,
                                                        float* __restrict__ C,
                                                        int M, int Nout, int K) {
  const int lane = threadIdx.x & 31;
  const int wv   = blockIdx.x * 8 + (threadIdx.x >> 5);
  const int ntn  = Nout >> 4;
  const int m0   = (wv / ntn) * 16;
  const int n0   = (wv % ntn) * 16;
  if (m0 >= M) return;            // wave-uniform
  v8f acc = {0.f,0.f,0.f,0.f,0.f,0.f,0.f,0.f};
  for (int k0 = 0; k0 < K; k0 += 32) {
    v16bf a = load_frag(A, K, m0, k0, lane);
    v16bf b = load_frag(W, K, n0, k0, lane);
    acc = wmma_bf16(a, b, acc);
  }
  const int mb = m0 + ((lane & 16) ? 8 : 0);
  const int n  = n0 + (lane & 15);
  const float bv = bias[n];
  #pragma unroll
  for (int r = 0; r < 8; ++r)
    C[(size_t)(mb + r) * Nout + n] = acc[r] + bv;
}

// ---------------- sequential main bi-GRU (2 blocks: fwd, bwd) ----------------

__global__ __launch_bounds__(512) void main_gru_kernel(const float* __restrict__ GiF,
                                                       const float* __restrict__ GiR,
                                                       const bf16_t* __restrict__ WhF,
                                                       const bf16_t* __restrict__ WhR,
                                                       const float* __restrict__ bhF,
                                                       const float* __restrict__ bhR,
                                                       bf16_t* __restrict__ outB) {
  __shared__ __align__(16) bf16_t hB[16 * 256];
  __shared__ float hF[8 * 256];
  __shared__ float gh[8 * G3];
  const int tid = threadIdx.x, lane = tid & 31, wave = tid >> 5;
  const int dir = blockIdx.x;                    // 0 = forward, 1 = reverse
  const float*  Gi = dir ? GiR : GiF;
  const bf16_t* Wh = dir ? WhR : WhF;
  const float*  bh = dir ? bhR : bhF;

  for (int e = tid; e < 16 * 256; e += 512) hB[e] = (bf16_t)0.f;
  for (int e = tid; e < 8 * 256;  e += 512) hF[e] = 0.f;
  __syncthreads();

  for (int t = 0; t < SEQ_N; ++t) {
    gh_gemm(hB, Wh, gh, wave, lane);
    __syncthreads();
    const int j = dir ? (SEQ_N - 1 - t) : t;     // input row / output row
    const float* gi = Gi + (size_t)j * BATCH * G3;
    float hv[4];
    #pragma unroll
    for (int u = 0; u < 4; ++u) {
      const int e = tid + u * 512, m = e >> 8, s = e & 255;
      const float* gim = gi + m * G3;
      const float* ghm = gh + m * G3;
      float r = sigm(gim[s]       + ghm[s]       + bh[s]);
      float z = sigm(gim[256 + s] + ghm[256 + s] + bh[256 + s]);
      float n = tanhf(gim[512 + s] + r * (ghm[512 + s] + bh[512 + s]));
      hv[u] = (1.f - z) * n + z * hF[e];
    }
    __syncthreads();
    #pragma unroll
    for (int u = 0; u < 4; ++u) {
      const int e = tid + u * 512, m = e >> 8, s = e & 255;
      hF[e] = hv[u];
      hB[e] = (bf16_t)hv[u];
      outB[((size_t)j * BATCH + m) * IN2H + dir * HID_H + s] = (bf16_t)hv[u];
    }
    __syncthreads();
  }
}

// ---------------- segment GRUs: 256 independent chains ----------------

__global__ __launch_bounds__(512) void seg_gru_kernel(const float* __restrict__ GiS,
                                                      const bf16_t* __restrict__ WhS,
                                                      const float* __restrict__ bh,
                                                      const float* __restrict__ h0f,
                                                      const float* __restrict__ h0b,
                                                      float* __restrict__ out) {
  __shared__ __align__(16) bf16_t hB[16 * 256];
  __shared__ float hF[8 * 256];
  __shared__ float gh[8 * G3];
  const int tid = threadIdx.x, lane = tid & 31, wave = tid >> 5;
  const int chain = blockIdx.x;                  // 0..255
  const int dir = chain >> 7;                    // 0 fwd, 1 bwd
  const int i0  = chain & 127;
  const float* h0 = (dir ? h0b : h0f) + (size_t)i0 * BATCH * SEG_S;

  for (int e = tid; e < 16 * 256; e += 512) {
    const int m = e >> 8;
    float v = (m < 8) ? h0[e] : 0.f;             // e == m*256+s for m<8
    hB[e] = (bf16_t)v;
    if (m < 8) hF[e] = v;
  }
  __syncthreads();

  const int tmax = dir ? ((SEG_T < i0 + 1) ? SEG_T : i0 + 1)
                       : ((SEG_T < SEQ_N - i0) ? SEG_T : SEQ_N - i0);
  for (int t = 0; t < tmax; ++t) {
    const int j = dir ? (i0 - t) : (i0 + t);
    gh_gemm(hB, WhS, gh, wave, lane);
    __syncthreads();
    const float* gi = GiS + (size_t)j * BATCH * G3;   // gi depends only on j!
    float hv[4];
    #pragma unroll
    for (int u = 0; u < 4; ++u) {
      const int e = tid + u * 512, m = e >> 8, s = e & 255;
      const float* gim = gi + m * G3;
      const float* ghm = gh + m * G3;
      float r = sigm(gim[s]       + ghm[s]       + bh[s]);
      float z = sigm(gim[256 + s] + ghm[256 + s] + bh[256 + s]);
      float n = tanhf(gim[512 + s] + r * (ghm[512 + s] + bh[512 + s]));
      hv[u] = (1.f - z) * n + z * hF[e];
    }
    __syncthreads();
    // triangular scatter: fwd -> (i0, j) lower half; bwd -> (j, i0) upper half
    const int rr = dir ? j : i0;
    const int cc = dir ? i0 : j;
    const long idx = (long)rr * SEQ_N - ((long)rr * (rr - 1)) / 2 + (cc - rr);
    float* dst = out + idx * (long)(BATCH * 2 * SEG_S) + dir * SEG_S;
    const bool wanthid = (i0 == SEQ_N - 1) && (dir ? (t == SEG_T - 1) : (t == 0));
    float* hid = out + (long)NPAIR * BATCH * 2 * SEG_S + dir * SEG_S;
    #pragma unroll
    for (int u = 0; u < 4; ++u) {
      const int e = tid + u * 512, m = e >> 8, s = e & 255;
      hF[e] = hv[u];
      hB[e] = (bf16_t)hv[u];
      dst[(size_t)m * (2 * SEG_S) + s] = hv[u];
      if (wanthid) hid[(size_t)m * (2 * SEG_S) + s] = hv[u];
    }
    __syncthreads();
  }
}

// ---------------- host launcher ----------------

extern "C" void kernel_launch(void* const* d_in, const int* in_sizes, int n_in,
                              void* d_out, int out_size, void* d_ws, size_t ws_size,
                              hipStream_t stream) {
  const int*   tokens = (const int*)d_in[0];
  const float* emb    = (const float*)d_in[1];
  const float* Wi_f   = (const float*)d_in[2];
  const float* Wh_f   = (const float*)d_in[3];
  const float* bi_f   = (const float*)d_in[4];
  const float* bh_f   = (const float*)d_in[5];
  const float* Wi_r   = (const float*)d_in[6];
  const float* Wh_r   = (const float*)d_in[7];
  const float* bi_r   = (const float*)d_in[8];
  const float* bh_r   = (const float*)d_in[9];
  const float* Wi_s   = (const float*)d_in[10];
  const float* Wh_s   = (const float*)d_in[11];
  const float* bi_s   = (const float*)d_in[12];
  const float* bh_s   = (const float*)d_in[13];
  const float* h0f    = (const float*)d_in[14];
  const float* h0b    = (const float*)d_in[15];
  float* out = (float*)d_out;

  char* ws = (char*)d_ws; size_t off = 0;
  auto take = [&](size_t nbytes) -> char* {
    char* p = ws + off; off += (nbytes + 255) & ~(size_t)255; return p;
  };
  bf16_t* Xb   = (bf16_t*)take((size_t)1024 * 256 * 2);
  bf16_t* WiFb = (bf16_t*)take((size_t)768 * 256 * 2);
  bf16_t* WhFb = (bf16_t*)take((size_t)768 * 256 * 2);
  bf16_t* WiRb = (bf16_t*)take((size_t)768 * 256 * 2);
  bf16_t* WhRb = (bf16_t*)take((size_t)768 * 256 * 2);
  bf16_t* WiSb = (bf16_t*)take((size_t)768 * 512 * 2);
  bf16_t* WhSb = (bf16_t*)take((size_t)768 * 256 * 2);
  bf16_t* OutB = (bf16_t*)take((size_t)1024 * 512 * 2);
  float*  GiF  = (float*)take((size_t)1024 * 768 * 4);
  float*  GiR  = (float*)take((size_t)1024 * 768 * 4);
  float*  GiS  = (float*)take((size_t)1024 * 768 * 4);

  // 1) zero the (mostly sparse) triangular output + hidden
  const int n4 = out_size / 4;
  zero_kernel<<<(n4 + 255) / 256, 256, 0, stream>>>((float4*)out, n4);

  // 2) bf16 weight copies (L2-resident thereafter)
  f2bf_kernel<<<768,  256, 0, stream>>>(Wi_f, WiFb, 768 * 256);
  f2bf_kernel<<<768,  256, 0, stream>>>(Wh_f, WhFb, 768 * 256);
  f2bf_kernel<<<768,  256, 0, stream>>>(Wi_r, WiRb, 768 * 256);
  f2bf_kernel<<<768,  256, 0, stream>>>(Wh_r, WhRb, 768 * 256);
  f2bf_kernel<<<1536, 256, 0, stream>>>(Wi_s, WiSb, 768 * 512);
  f2bf_kernel<<<768,  256, 0, stream>>>(Wh_s, WhSb, 768 * 256);

  // 3) embedding gather -> bf16 X (1024 x 256)
  embed_kernel<<<1024, 256, 0, stream>>>(tokens, emb, Xb);

  // 4) hoisted input projections for both main GRUs: Gi = X @ Wi^T + bi
  gemm_bias_kernel<<<384, 256, 0, stream>>>(Xb, WiFb, bi_f, GiF, 1024, G3, 256);
  gemm_bias_kernel<<<384, 256, 0, stream>>>(Xb, WiRb, bi_r, GiR, 1024, G3, 256);

  // 5) sequential bidirectional GRU (2 independent chains)
  main_gru_kernel<<<2, 512, 0, stream>>>(GiF, GiR, WhFb, WhRb, bh_f, bh_r, OutB);

  // 6) hoisted segment input projection: only 128 distinct gi_s rows exist
  gemm_bias_kernel<<<384, 256, 0, stream>>>(OutB, WiSb, bi_s, GiS, 1024, G3, 512);

  // 7) 256 independent segment chains -> triangular output + hidden
  seg_gru_kernel<<<256, 512, 0, stream>>>(GiS, WhSb, bh_s, h0f, h0b, out);
}
